// BaseEncoderBiRNN_89000312308179
// MI455X (gfx1250) — matrix operand order repaired and checked by reading, not compiled, not run on hardware
//
#include <hip/hip_runtime.h>
#include <hip/hip_bf16.h>

// ---------------- problem constants ----------------
#define HDIM 1024
#define LSEQ 4096
#define G3H  (3 * HDIM)          // 3072
#define NWG  64                  // workgroups per direction in the recurrence
#define CH   (HDIM / NWG)        // 16 h-elements owned per WG
#define ROWS (3 * CH)            // 48 Whh rows per WG per step

// ---------------- WMMA vector types ----------------
typedef __attribute__((ext_vector_type(16))) __bf16 v16bf;
typedef __attribute__((ext_vector_type(8)))  __bf16 bf16x8;
typedef __attribute__((ext_vector_type(8)))  float  v8f;

union AFrag { v16bf v; bf16x8 h[2]; };

// ---------------- ws layout ----------------
// [Gi_fwd f32 L*3H][Gi_rev f32 L*3H]
// [xb_hi bf16 L*H][xb_lo bf16 L*H]
// [wb_f_hi][wb_f_lo][wb_r_hi][wb_r_lo]  (each bf16 3H*H)
// [h f32 2*H][cnt i32 2*L]
static inline size_t off_gi(int dir)          { return (size_t)dir * LSEQ * G3H * 4; }
static inline size_t off_xb(int part)         { return (size_t)2 * LSEQ * G3H * 4 + (size_t)part * LSEQ * HDIM * 2; }
static inline size_t off_wb(int dir, int part){ return off_xb(2) + ((size_t)(2 * dir + part)) * G3H * HDIM * 2; }
static inline size_t off_h()                  { return off_wb(2, 0); }
static inline size_t off_cnt()                { return off_h() + (size_t)2 * HDIM * 4; }

// ---------------- init: zero h + barrier counters ----------------
__global__ void gru_init_kernel(float* h, int* cnt) {
    int idx = blockIdx.x * blockDim.x + threadIdx.x;
    if (idx < 2 * LSEQ) cnt[idx] = 0;
    if (idx < 2 * HDIM) h[idx] = 0.0f;
}

// ---------------- fp32 -> split bf16 (hi + lo) for x and both Wih ----------------
__global__ void gru_cvt_kernel(const float* __restrict__ x,
                               const float* __restrict__ wf,
                               const float* __restrict__ wr,
                               __bf16* __restrict__ xhi, __bf16* __restrict__ xlo,
                               __bf16* __restrict__ wfhi, __bf16* __restrict__ wflo,
                               __bf16* __restrict__ wrhi, __bf16* __restrict__ wrlo) {
    size_t idx = (size_t)blockIdx.x * blockDim.x + threadIdx.x;
    if (idx < (size_t)LSEQ * HDIM) {
        float v = x[idx];
        __bf16 hi = (__bf16)v;
        xhi[idx] = hi;
        xlo[idx] = (__bf16)(v - (float)hi);
    }
    if (idx < (size_t)G3H * HDIM) {
        float v = wf[idx];
        __bf16 hi = (__bf16)v;
        wfhi[idx] = hi;
        wflo[idx] = (__bf16)(v - (float)hi);
        v = wr[idx];
        hi = (__bf16)v;
        wrhi[idx] = hi;
        wrlo[idx] = (__bf16)(v - (float)hi);
    }
}

// ---------------- Gi = X @ Wih^T + bih  (split-bf16 WMMA, fp32 accum) ----------------
// One 16x16 output tile per wave; 8 waves per block. Per K-step of 32:
//   acc += Ahi*Bhi + Ahi*Blo + Alo*Bhi   (3 x v_wmma_f32_16x16x32_bf16)
__global__ void gru_gemm_kernel(const __bf16* __restrict__ xhi,
                                const __bf16* __restrict__ xlo,
                                const __bf16* __restrict__ wb_f_hi,
                                const __bf16* __restrict__ wb_f_lo,
                                const __bf16* __restrict__ wb_r_hi,
                                const __bf16* __restrict__ wb_r_lo,
                                const float* __restrict__ bih_f,
                                const float* __restrict__ bih_r,
                                float* __restrict__ gi_f,
                                float* __restrict__ gi_r) {
    const int wave   = threadIdx.x >> 5;
    const int lane   = threadIdx.x & 31;
    const int tileId = blockIdx.x * 8 + wave;

    const int tilesPerDir = (LSEQ / 16) * (G3H / 16);   // 49152
    const int dir   = tileId / tilesPerDir;
    const int tid2  = tileId - dir * tilesPerDir;
    const int tileM = tid2 / (G3H / 16);
    const int tileN = tid2 - tileM * (G3H / 16);
    const int tbase = tileM * 16;
    const int nbase = tileN * 16;

    const __bf16* wbhi = dir ? wb_r_hi : wb_f_hi;
    const __bf16* wblo = dir ? wb_r_lo : wb_f_lo;
    const float*  bih  = dir ? bih_r : bih_f;
    float*        gi   = dir ? gi_r : gi_f;

    const int m  = lane & 15;
    const int kh = lane >> 4;

    const __bf16* aHiRow = xhi + (size_t)(tbase + m) * HDIM;
    const __bf16* aLoRow = xlo + (size_t)(tbase + m) * HDIM;
    const __bf16* bHiRow = wbhi + (size_t)(nbase + m) * HDIM;
    const __bf16* bLoRow = wblo + (size_t)(nbase + m) * HDIM;

    v8f acc = {};
    #pragma unroll 2
    for (int k = 0; k < HDIM; k += 32) {
        AFrag ah, al;
        ah.v = *(const v16bf*)(aHiRow + k + 16 * kh);
        al.v = *(const v16bf*)(aLoRow + k + 16 * kh);
        v16bf bh = *(const v16bf*)(bHiRow + k + 16 * kh);
        v16bf bl = *(const v16bf*)(bLoRow + k + 16 * kh);
        acc = __builtin_amdgcn_wmma_f32_16x16x32_bf16(
            false, ah.v, false, bh, (short)0, acc, false, false);
        acc = __builtin_amdgcn_wmma_f32_16x16x32_bf16(
            false, ah.v, false, bl, (short)0, acc, false, false);
        acc = __builtin_amdgcn_wmma_f32_16x16x32_bf16(
            false, al.v, false, bh, (short)0, acc, false, false);
    }

    const int   n    = nbase + m;
    const float bias = bih[n];
    #pragma unroll
    for (int r = 0; r < 8; ++r) {
        int row = tbase + r + 8 * kh;           // C layout: lanes 16-31 hold M = r+8
        gi[(size_t)row * G3H + n] = acc[r] + bias;
    }
}

// ---------------- persistent recurrence kernel ----------------
// grid = 2*NWG blocks of 256 threads. Block (dir, wg) owns h[jbase .. jbase+CH).
// Per step: async-stage h into LDS (global_load_async_to_lds_b128, ASYNCcnt),
// wave-per-row fp32 dot products against Whh (L2 resident), shuffle reduce,
// gate math, write h + outputs, grid barrier. Next step's Gi entries are
// prefetched (global_prefetch_b8) while the dots run.
__global__ void gru_rec_kernel(const float* __restrict__ gi_f,
                               const float* __restrict__ gi_r,
                               const float* __restrict__ whh_f,
                               const float* __restrict__ whh_r,
                               const float* __restrict__ bhh_f,
                               const float* __restrict__ bhh_r,
                               float* __restrict__ hglob,
                               int* __restrict__ cnt,
                               float* __restrict__ out) {
    const int dir = blockIdx.x >> 6;            // NWG == 64
    const int wg  = blockIdx.x & (NWG - 1);

    const float* gi  = dir ? gi_r : gi_f;
    const float* whh = dir ? whh_r : whh_f;
    const float* bhh = dir ? bhh_r : bhh_f;
    float*       hg  = hglob + dir * HDIM;
    int*         cn  = cnt + dir * LSEQ;

    __shared__ __align__(16) float hs[HDIM];
    __shared__ float ghs[ROWS];

    const int jbase = wg * CH;
    const int wave  = threadIdx.x >> 5;
    const int lane  = threadIdx.x & 31;

    // per-thread addresses for async h staging: 256 threads x 16B = 4KB = h
    const float*   hsrc   = hg + 4 * threadIdx.x;
    const unsigned hs_off = (unsigned)(size_t)&hs[4 * threadIdx.x];

    for (int t = 0; t < LSEQ; ++t) {
        const int t_x = dir ? (LSEQ - 1 - t) : t;

        // async-stage full h into LDS (memory -> LDS direct, no VGPR roundtrip)
        asm volatile("global_load_async_to_lds_b128 %0, %1, off"
                     :
                     : "v"(hs_off), "v"(hsrc)
                     : "memory");
        asm volatile("s_wait_asynccnt 0x0" ::: "memory");
        __syncthreads();

        // prefetch next step's Gi entries while the dots run
        if (t + 1 < LSEQ && threadIdx.x < CH) {
            const int tn = dir ? (LSEQ - 2 - t) : (t + 1);
            const float* gnext = gi + (size_t)tn * G3H;
            const int j = jbase + threadIdx.x;
            __builtin_prefetch(&gnext[j], 0, 1);
            __builtin_prefetch(&gnext[HDIM + j], 0, 1);
            __builtin_prefetch(&gnext[2 * HDIM + j], 0, 1);
        }

        // wave-per-row dot products: rows g*H + jbase + i, g in {r,z,n}
        const float4* hv4 = (const float4*)hs;
        for (int rr = wave; rr < ROWS; rr += 8) {
            const int g = rr / CH;
            const int i = rr - g * CH;
            const int nrow = g * HDIM + jbase + i;
            const float4* wrow = (const float4*)(whh + (size_t)nrow * HDIM);
            float acc = 0.0f;
            #pragma unroll 2
            for (int q = lane; q < HDIM / 4; q += 32) {
                float4 w  = wrow[q];
                float4 hh = hv4[q];
                acc += w.x * hh.x + w.y * hh.y + w.z * hh.z + w.w * hh.w;
            }
            #pragma unroll
            for (int offv = 16; offv > 0; offv >>= 1)
                acc += __shfl_xor(acc, offv, 32);
            if (lane == 0) ghs[rr] = acc;
        }
        __syncthreads();

        // gate math + state update for owned h elements
        if (threadIdx.x < CH) {
            const int j = jbase + threadIdx.x;
            const float* girow = gi + (size_t)t_x * G3H;
            float ir  = girow[j];
            float iz  = girow[HDIM + j];
            float inn = girow[2 * HDIM + j];
            float hr = ghs[threadIdx.x]          + bhh[j];
            float hz = ghs[CH + threadIdx.x]     + bhh[HDIM + j];
            float hn = ghs[2 * CH + threadIdx.x] + bhh[2 * HDIM + j];
            float r  = 1.0f / (1.0f + expf(-(ir + hr)));
            float z  = 1.0f / (1.0f + expf(-(iz + hz)));
            float n  = tanhf(inn + r * hn);
            float hnew = (1.0f - z) * n + z * hs[j];
            hg[j] = hnew;
            out[(size_t)t_x * (2 * HDIM) + dir * HDIM + j] = hnew;
        }
        __syncthreads();

        // per-direction grid barrier (step-indexed counters, reset by init)
        __threadfence();
        if (threadIdx.x == 0) {
            atomicAdd(&cn[t], 1);
            while (__hip_atomic_load(&cn[t], __ATOMIC_ACQUIRE,
                                     __HIP_MEMORY_SCOPE_AGENT) < NWG)
                __builtin_amdgcn_s_sleep(1);
        }
        __syncthreads();
        __threadfence();
    }

    // final hidden state + projected copy
    if (threadIdx.x < CH) {
        const int j = jbase + threadIdx.x;
        float hv = hg[j];
        out[(size_t)LSEQ * 2 * HDIM + dir * HDIM + j] = hv;
        out[(size_t)LSEQ * 2 * HDIM + 2 * HDIM + dir * HDIM + j] = hv;
    }
}

extern "C" void kernel_launch(void* const* d_in, const int* in_sizes, int n_in,
                              void* d_out, int out_size, void* d_ws, size_t ws_size,
                              hipStream_t stream) {
    const float* x       = (const float*)d_in[0];
    const float* fwd_Wih = (const float*)d_in[1];
    const float* fwd_Whh = (const float*)d_in[2];
    const float* fwd_bih = (const float*)d_in[3];
    const float* fwd_bhh = (const float*)d_in[4];
    const float* rev_Wih = (const float*)d_in[5];
    const float* rev_Whh = (const float*)d_in[6];
    const float* rev_bih = (const float*)d_in[7];
    const float* rev_bhh = (const float*)d_in[8];
    float* out = (float*)d_out;

    char* ws = (char*)d_ws;
    float*  gi_f  = (float*)(ws + off_gi(0));
    float*  gi_r  = (float*)(ws + off_gi(1));
    __bf16* xhi   = (__bf16*)(ws + off_xb(0));
    __bf16* xlo   = (__bf16*)(ws + off_xb(1));
    __bf16* wfhi  = (__bf16*)(ws + off_wb(0, 0));
    __bf16* wflo  = (__bf16*)(ws + off_wb(0, 1));
    __bf16* wrhi  = (__bf16*)(ws + off_wb(1, 0));
    __bf16* wrlo  = (__bf16*)(ws + off_wb(1, 1));
    float*  hbuf  = (float*)(ws + off_h());
    int*    cnt   = (int*)(ws + off_cnt());

    // 1) zero h and barrier counters (every call -> deterministic graph replay)
    gru_init_kernel<<<(2 * LSEQ + 255) / 256, 256, 0, stream>>>(hbuf, cnt);

    // 2) fp32 -> split bf16 (hi/lo) for x and both Wih
    {
        size_t total = (size_t)LSEQ * HDIM;
        gru_cvt_kernel<<<(unsigned)((total + 255) / 256), 256, 0, stream>>>(
            x, fwd_Wih, rev_Wih, xhi, xlo, wfhi, wflo, wrhi, wrlo);
    }

    // 3) split-bf16 WMMA GEMM: Gi = X @ Wih^T + bih, both directions
    {
        const int tiles = 2 * (LSEQ / 16) * (G3H / 16);   // 98304
        gru_gemm_kernel<<<tiles / 8, 256, 0, stream>>>(
            xhi, xlo, wfhi, wflo, wrhi, wrlo, fwd_bih, rev_bih, gi_f, gi_r);
    }

    // 4) persistent bidirectional recurrence
    gru_rec_kernel<<<2 * NWG, 256, 0, stream>>>(
        gi_f, gi_r, fwd_Whh, rev_Whh, fwd_bhh, rev_bhh, hbuf, cnt, out);
}
